// VAE_66769561583724
// MI455X (gfx1250) — matrix operand order repaired and checked by reading, not compiled, not run on hardware
//
#include <hip/hip_runtime.h>
#include <hip/hip_bf16.h>
#include <math.h>

// ---------------------------------------------------------------------------
// CDNA5 WMMA types (wave32): v_wmma_f32_16x16x32_f16
// ---------------------------------------------------------------------------
typedef _Float16 half16 __attribute__((ext_vector_type(16)));
typedef float    float8 __attribute__((ext_vector_type(8)));

#define VAE_DIMZ  100
#define VAE_B     4096

#if __has_builtin(__builtin_amdgcn_global_load_async_to_lds_b128)
#define VAE_ASYNC_LDS 1
typedef int vae_i4 __attribute__((__vector_size__(16)));   // matches builtin's param type
#endif

// ---------------------------------------------------------------------------
// Weight pre-pack: W_packed[co][kg] (f16, zero-padded to Kpad) with kg =
// (ci,ky,kx) flattened; handles conv (OIHW) vs convT (IOHW) indexing once.
// ---------------------------------------------------------------------------
__global__ void pack_w_kernel(const float* __restrict__ w,
                              _Float16* __restrict__ wp,
                              int Cout, int Cin, int k, int Kdim, int Kpad,
                              int mode) {
  int idx = blockIdx.x * 256 + threadIdx.x;
  if (idx >= Cout * Kpad) return;
  int co = idx / Kpad;
  int kk = idx - co * Kpad;
  float v = 0.0f;
  if (kk < Kdim) {
    int kk2 = k * k;
    int ci = kk / kk2;
    int r  = kk - ci * kk2;
    int ky = r / k;
    int kx = r - ky * k;
    v = (mode == 0) ? w[((co * Cin + ci) * k + ky) * k + kx]
                    : w[((ci * Cout + co) * k + ky) * k + kx];
  }
  wp[idx] = (_Float16)v;
}

// ---------------------------------------------------------------------------
// Implicit-GEMM conv / convT / linear.
//   M = Cout (16/block), N = batch*OH*OW (256/block; 32/wave -> 2 WMMA tiles),
//   K = Cin*k*k stepped by 32.
// ---------------------------------------------------------------------------
struct ConvP {
  const float*    __restrict__ x;
  const _Float16* __restrict__ wp;   // packed [Cout][Kpad] f16
  const float*    __restrict__ b;
  float*          __restrict__ y;
  int Hin, Win, Cout, Hout, Wout;
  int k, stride, pad;
  int isn, isc, isy, isx;            // input element strides (NHWC-capable)
  int Kdim, Kpad, Npos;
  int mode;                          // 0 = conv, 1 = convT (gather)
  int relu;
};

__global__ __launch_bounds__(256) void conv_wmma_kernel(ConvP p) {
  __shared__ _Float16 ldsA[16 * 32];        // weight tile [m][k]
  __shared__ _Float16 ldsB[8][32 * 32];     // per-wave im2col tile [n][k]

  const int tid   = threadIdx.x;
  const int lane  = tid & 31;
  const int wid   = tid >> 5;
  const int mbase = blockIdx.y << 4;
  const int nbase = (blockIdx.x << 8) + (wid << 5);

  const int HW = p.Hout * p.Wout;

  // ---- per-lane B-row precompute: pos -> (n,oy,ox), per-tap offsets ----
  const int  bpos   = nbase + lane;           // this lane owns row n = lane
  const bool bvalid = bpos < p.Npos;
  int baseN = 0;
  int yoff[4], xoff[4];
  bool vy[4], vx[4];
  {
    int n = 0, oy = 0, ox = 0;
    if (bvalid) {
      n = bpos / HW;
      int r = bpos - n * HW;
      oy = r / p.Wout;
      ox = r - oy * p.Wout;
    }
    baseN = n * p.isn;
    for (int t = 0; t < 4; ++t) { yoff[t] = 0; xoff[t] = 0; vy[t] = false; vx[t] = false; }
    for (int t = 0; t < p.k; ++t) {
      int iy, ix; bool oky, okx;
      if (p.mode == 0) {
        iy = oy * p.stride - p.pad + t;  oky = (iy >= 0) && (iy < p.Hin);
        ix = ox * p.stride - p.pad + t;  okx = (ix >= 0) && (ix < p.Win);
      } else {
        int ny = oy + p.pad - t;  iy = ny / p.stride;
        oky = (ny >= 0) && (iy * p.stride == ny) && (iy < p.Hin);
        int nx = ox + p.pad - t;  ix = nx / p.stride;
        okx = (nx >= 0) && (ix * p.stride == nx) && (ix < p.Win);
      }
      vy[t] = oky; yoff[t] = oky ? iy * p.isy : 0;
      vx[t] = okx; xoff[t] = okx ? ix * p.isx : 0;
    }
  }

#ifdef VAE_ASYNC_LDS
  // Pre-zero A tile: rows with co >= Cout are never written by the async path.
  ((unsigned int*)ldsA)[tid & 255] = 0u;
#endif

  float8 acc0 = {};
  float8 acc1 = {};

  // carry-propagated (ci,ky,kx) counters — no division in the hot loop
  int ci = 0, ky = 0, kx = 0;

  for (int k0 = 0; k0 < p.Kdim; k0 += 32) {
    __syncthreads();   // previous iteration's LDS reads complete (and A zero-fill)

    // ---- stage A (packed weights, contiguous) ----
#ifdef VAE_ASYNC_LDS
    if (wid == 0) {
      // two async B128 issues cover the 16x32-half (1 KB) tile
      int m0 = lane >> 2;                 // rows 0..7
      int c0 = (lane & 3) << 3;           // 8-half (16B) chunk
      if (mbase + m0 < p.Cout) {
        const _Float16* g = p.wp + (size_t)(mbase + m0) * p.Kpad + k0 + c0;
        __builtin_amdgcn_global_load_async_to_lds_b128(
            (vae_i4*)g, (vae_i4*)&ldsA[m0 * 32 + c0], 0, 0);
      }
      int m1 = m0 + 8;                    // rows 8..15
      if (mbase + m1 < p.Cout) {
        const _Float16* g = p.wp + (size_t)(mbase + m1) * p.Kpad + k0 + c0;
        __builtin_amdgcn_global_load_async_to_lds_b128(
            (vae_i4*)g, (vae_i4*)&ldsA[m1 * 32 + c0], 0, 0);
      }
    }
#else
    {
      int am  = tid >> 4;
      int ak0 = (tid & 15) << 1;
      int co  = mbase + am;
      unsigned int wv = 0u;
      if (co < p.Cout)
        wv = *(const unsigned int*)(p.wp + (size_t)co * p.Kpad + k0 + ak0);
      ((unsigned int*)ldsA)[am * 16 + (ak0 >> 1)] = wv;
    }
#endif

    // ---- stage B (fused im2col: predicate + add + load, no division) ----
    #pragma unroll 8
    for (int j = 0; j < 32; ++j) {
      int   kg = k0 + j;
      float v  = 0.0f;
      if (bvalid && kg < p.Kdim && vy[ky] && vx[kx])
        v = p.x[baseN + ci * p.isc + yoff[ky] + xoff[kx]];
      ldsB[wid][lane * 32 + j] = (_Float16)v;
      ++kx;
      if (kx == p.k) { kx = 0; ++ky; if (ky == p.k) { ky = 0; ++ci; } }
    }

#ifdef VAE_ASYNC_LDS
    if (wid == 0) {
#if __has_builtin(__builtin_amdgcn_s_wait_asynccnt)
      __builtin_amdgcn_s_wait_asynccnt(0);
#else
      asm volatile("s_wait_asynccnt 0" ::: "memory");
#endif
    }
#endif
    __syncthreads();   // tiles visible to all waves

    // ---- fragments: contiguous k = 16*(lane/16) + s in both A and B ----
    const unsigned int* A32 = (const unsigned int*)ldsA;
    const unsigned int* B32 = (const unsigned int*)&ldsB[wid][0];
    const int hi = lane >> 4;
    const int ml = lane & 15;

    union { unsigned int u[8]; half16 h; } af, bf0, bf1;
    // A 16x32: lanes0-15 hold K=0..15 (m = lane), lanes16-31 hold K=16..31
    const int akb = hi << 3;
    #pragma unroll
    for (int v = 0; v < 4; ++v) af.u[v]     = A32[ml * 16 + akb + v];
    #pragma unroll
    for (int v = 0; v < 4; ++v) af.u[4 + v] = A32[ml * 16 + 4 + akb + v];
    // B 32x16: lanes0-15 hold K=0..15 (n = lane), lanes16-31 hold K=16..31
    const int bkb = hi << 3;
    #pragma unroll
    for (int v = 0; v < 8; ++v) bf0.u[v] = B32[(ml)      * 16 + bkb + v];
    #pragma unroll
    for (int v = 0; v < 8; ++v) bf1.u[v] = B32[(16 + ml) * 16 + bkb + v];

    acc0 = __builtin_amdgcn_wmma_f32_16x16x32_f16(false, af.h, false, bf0.h,
                                                  (short)0, acc0, false, false);
    acc1 = __builtin_amdgcn_wmma_f32_16x16x32_f16(false, af.h, false, bf1.h,
                                                  (short)0, acc1, false, false);
  }

  // ---- epilogue: bias + relu + store (D: lanes0-15 N=lane,M=r; lanes16-31 N=lane-16,M=8+r)
  const int hi = lane >> 4;
  const int nn = lane & 15;
  #pragma unroll
  for (int t = 0; t < 2; ++t) {
    int pos = nbase + (t << 4) + nn;
    if (pos < p.Npos) {
      int n  = pos / HW;
      int r  = pos - n * HW;
      int oy = r / p.Wout;
      int ox = r - oy * p.Wout;
      const float8& acc = t ? acc1 : acc0;
      #pragma unroll
      for (int rr = 0; rr < 8; ++rr) {
        int co = mbase + (hi << 3) + rr;
        if (co < p.Cout) {
          float v = acc[rr] + p.b[co];
          if (p.relu) v = fmaxf(v, 0.0f);
          p.y[((n * p.Cout + co) * p.Hout + oy) * p.Wout + ox] = v;
        }
      }
    }
  }
}

// ---------------------------------------------------------------------------
// PoE fusion + reparameterization
// ---------------------------------------------------------------------------
__device__ __forceinline__ float softplus_f(float x) {
  return fmaxf(x, 0.0f) + log1pf(expf(-fabsf(x)));
}

__global__ void fuse_kernel(const float* __restrict__ z1,
                            const float* __restrict__ z2,
                            const float* __restrict__ eps,
                            float* __restrict__ mu_out,
                            float* __restrict__ var_out,
                            float* __restrict__ sample) {
  int i = blockIdx.x * blockDim.x + threadIdx.x;
  if (i >= VAE_B * VAE_DIMZ) return;
  int n = i / VAE_DIMZ;
  int d = i - n * VAE_DIMZ;
  int rb = n * (2 * VAE_DIMZ);
  float mu1 = z1[rb + d];
  float mu2 = z2[rb + d];
  float v1  = softplus_f(0.5f * z1[rb + VAE_DIMZ + d]);
  float v2  = softplus_f(0.5f * z2[rb + VAE_DIMZ + d]);
  float s   = v1 + v2;
  float mu  = (mu1 * v1 + mu2 * v2) / s;
  float vz  = 1.0f / s;
  mu_out[i]  = mu;
  var_out[i] = vz;
  sample[i]  = mu + eps[i] * sqrtf(vz);
}

// ---------------------------------------------------------------------------
// Host side
// ---------------------------------------------------------------------------
static const size_t kPSz[48] = {
  512, 32, 16384, 32, 32768, 64, 65536, 64, 16384, 256,        // me_w/b 1..5
  1536, 32, 16384, 32, 32768, 64, 65536, 64, 65536, 256,       // se_w/b 1..5
  51200, 200, 51200, 200,                                      // lm, ls
  25600, 256, 25600, 256,                                      // dl1, dl2
  65536, 64, 36864, 64, 32768, 32, 16384, 32, 128, 1,          // md_w/b 1..5
  65536, 64, 65536, 64, 32768, 32, 16384, 32, 1536, 3          // sd_w/b 1..5
};

static void run_layer(hipStream_t st,
                      const float* x, const float* w, const float* b, float* y,
                      _Float16* wp,
                      int N, int Cin, int Hin, int Win,
                      int Cout, int Hout, int Wout,
                      int k, int stride, int pad, int mode, int relu,
                      int isn, int isc, int isy, int isx) {
  int Kdim = Cin * k * k;
  int Kpad = (Kdim + 31) & ~31;
  int tot  = Cout * Kpad;
  pack_w_kernel<<<(tot + 255) / 256, 256, 0, st>>>(w, wp, Cout, Cin, k, Kdim, Kpad, mode);

  ConvP p;
  p.x = x; p.wp = wp; p.b = b; p.y = y;
  p.Hin = Hin; p.Win = Win; p.Cout = Cout; p.Hout = Hout; p.Wout = Wout;
  p.k = k; p.stride = stride; p.pad = pad;
  p.isn = isn; p.isc = isc; p.isy = isy; p.isx = isx;
  p.Kdim = Kdim; p.Kpad = Kpad; p.Npos = N * Hout * Wout;
  p.mode = mode; p.relu = relu;
  dim3 grid((p.Npos + 255) / 256, (Cout + 15) / 16);
  conv_wmma_kernel<<<grid, dim3(256), 0, st>>>(p);
}

static void run_layer_nchw(hipStream_t st,
                           const float* x, const float* w, const float* b, float* y,
                           _Float16* wp,
                           int N, int Cin, int Hin, int Win,
                           int Cout, int Hout, int Wout,
                           int k, int stride, int pad, int mode, int relu) {
  run_layer(st, x, w, b, y, wp, N, Cin, Hin, Win, Cout, Hout, Wout,
            k, stride, pad, mode, relu,
            Cin * Hin * Win, Hin * Win, Win, 1);
}

extern "C" void kernel_launch(void* const* d_in, const int* in_sizes, int n_in,
                              void* d_out, int out_size, void* d_ws, size_t ws_size,
                              hipStream_t stream) {
  const float* x1  = (const float*)d_in[0];   // (B,28,28)
  const float* x2  = (const float*)d_in[1];   // (B,32,32,3) NHWC
  const float* eps = (const float*)d_in[2];   // (B,100)

  auto P = [&](int j) -> const float* {
    if (n_in >= 51) return (const float*)d_in[3 + j];
    const float* base = (const float*)d_in[3];
    size_t off = 0;
    for (int i = 0; i < j; ++i) off += kPSz[i];
    return base + off;
  };
  auto ME_W = [&](int i){ return P((i - 1) * 2); };
  auto ME_B = [&](int i){ return P((i - 1) * 2 + 1); };
  auto SE_W = [&](int i){ return P(10 + (i - 1) * 2); };
  auto SE_B = [&](int i){ return P(10 + (i - 1) * 2 + 1); };
  const float *LM_W = P(20), *LM_B = P(21), *LS_W = P(22), *LS_B = P(23);
  const float *DL1_W = P(24), *DL1_B = P(25), *DL2_W = P(26), *DL2_B = P(27);
  auto MD_W = [&](int i){ return P(28 + (i - 1) * 2); };
  auto MD_B = [&](int i){ return P(28 + (i - 1) * 2 + 1); };
  auto SD_W = [&](int i){ return P(38 + (i - 1) * 2); };
  auto SD_B = [&](int i){ return P(38 + (i - 1) * 2 + 1); };

  // Workspace: 2 ping-pong activation buffers + latents + packed-weight scratch.
  float* ws = (float*)d_ws;
  const size_t CAP = 33554432;               // 4096*32*16*16 floats
  float* bufA = ws;
  float* bufB = ws + CAP;
  float* z1   = ws + 2 * CAP;                                  // 4096*200
  float* z2   = z1 + (size_t)VAE_B * 200;
  float* smp  = z2 + (size_t)VAE_B * 200;                      // 4096*100
  _Float16* wp = (_Float16*)(smp + (size_t)VAE_B * VAE_DIMZ);  // 65536 halves max

  float* out    = (float*)d_out;
  float* outD1  = out;                       // 4096*1*28*28
  float* outD2  = out + 3211264;             // 4096*3*32*32
  float* outMU  = out + 15794176;            // 4096*100
  float* outVAR = out + 16203776;            // 4096*100

  const int N = VAE_B;

  // ---------------- Encoder 1 (MNIST 1x28x28) ----------------
  run_layer(stream, x1, ME_W(1), ME_B(1), bufA, wp, N, 1, 28, 28, 32, 14, 14,
            4, 2, 1, 0, 1, 784, 784, 28, 1);
  run_layer_nchw(stream, bufA, ME_W(2), ME_B(2), bufB, wp, N, 32, 14, 14, 32, 7, 7, 4, 2, 1, 0, 1);
  run_layer_nchw(stream, bufB, ME_W(3), ME_B(3), bufA, wp, N, 32, 7, 7, 64, 3, 3, 4, 2, 1, 0, 1);
  run_layer_nchw(stream, bufA, ME_W(4), ME_B(4), bufB, wp, N, 64, 3, 3, 64, 1, 1, 4, 2, 1, 0, 1);
  run_layer_nchw(stream, bufB, ME_W(5), ME_B(5), bufA, wp, N, 64, 1, 1, 256, 1, 1, 1, 1, 0, 0, 1);
  run_layer_nchw(stream, bufA, LM_W, LM_B, z1, wp, N, 256, 1, 1, 200, 1, 1, 1, 1, 0, 0, 0);

  // ---------------- Encoder 2 (SVHN 3x32x32, NHWC) ----------------
  run_layer(stream, x2, SE_W(1), SE_B(1), bufA, wp, N, 3, 32, 32, 32, 16, 16,
            4, 2, 1, 0, 1, 3072, 1, 96, 3);
  run_layer_nchw(stream, bufA, SE_W(2), SE_B(2), bufB, wp, N, 32, 16, 16, 32, 8, 8, 4, 2, 1, 0, 1);
  run_layer_nchw(stream, bufB, SE_W(3), SE_B(3), bufA, wp, N, 32, 8, 8, 64, 4, 4, 4, 2, 1, 0, 1);
  run_layer_nchw(stream, bufA, SE_W(4), SE_B(4), bufB, wp, N, 64, 4, 4, 64, 2, 2, 4, 2, 1, 0, 1);
  run_layer_nchw(stream, bufB, SE_W(5), SE_B(5), bufA, wp, N, 64, 2, 2, 256, 1, 1, 2, 1, 0, 0, 1);
  run_layer_nchw(stream, bufA, LS_W, LS_B, z2, wp, N, 256, 1, 1, 200, 1, 1, 1, 1, 0, 0, 0);

  // ---------------- PoE fusion + sampling ----------------
  {
    int tot = N * VAE_DIMZ;
    fuse_kernel<<<(tot + 255) / 256, 256, 0, stream>>>(z1, z2, eps, outMU, outVAR, smp);
  }

  // ---------------- Decoder 1 (MNIST) ----------------
  run_layer_nchw(stream, smp, DL1_W, DL1_B, bufA, wp, N, 100, 1, 1, 256, 1, 1, 1, 1, 0, 0, 1);
  run_layer_nchw(stream, bufA, MD_W(1), MD_B(1), bufB, wp, N, 256, 1, 1, 64, 2, 2, 2, 1, 0, 1, 1);
  run_layer_nchw(stream, bufB, MD_W(2), MD_B(2), bufA, wp, N, 64, 2, 2, 64, 3, 3, 3, 2, 1, 1, 1);
  run_layer_nchw(stream, bufA, MD_W(3), MD_B(3), bufB, wp, N, 64, 3, 3, 32, 6, 6, 4, 2, 1, 1, 1);
  run_layer_nchw(stream, bufB, MD_W(4), MD_B(4), bufA, wp, N, 32, 6, 6, 32, 14, 14, 4, 2, 0, 1, 1);
  run_layer_nchw(stream, bufA, MD_W(5), MD_B(5), outD1, wp, N, 32, 14, 14, 1, 28, 28, 2, 2, 0, 1, 1);

  // ---------------- Decoder 2 (SVHN) ----------------
  run_layer_nchw(stream, smp, DL2_W, DL2_B, bufA, wp, N, 100, 1, 1, 256, 1, 1, 1, 1, 0, 0, 1);
  run_layer_nchw(stream, bufA, SD_W(1), SD_B(1), bufB, wp, N, 256, 1, 1, 64, 2, 2, 2, 1, 0, 1, 1);
  run_layer_nchw(stream, bufB, SD_W(2), SD_B(2), bufA, wp, N, 64, 2, 2, 64, 4, 4, 4, 2, 1, 1, 1);
  run_layer_nchw(stream, bufA, SD_W(3), SD_B(3), bufB, wp, N, 64, 4, 4, 32, 8, 8, 4, 2, 1, 1, 1);
  run_layer_nchw(stream, bufB, SD_W(4), SD_B(4), bufA, wp, N, 32, 8, 8, 32, 16, 16, 4, 2, 1, 1, 1);
  run_layer_nchw(stream, bufA, SD_W(5), SD_B(5), outD2, wp, N, 32, 16, 16, 3, 32, 32, 4, 2, 1, 1, 1);
}
